// GCNClassifier_19344532701419
// MI455X (gfx1250) — matrix operand order repaired and checked by reading, not compiled, not run on hardware
//
#include <hip/hip_runtime.h>
#include <hip/hip_bf16.h>

// Problem constants (from reference)
#define NNODES 50000
#define NEDGES 1600000
#define NGRAPH 512
#define HID    128
#define CIN    3
#define CMID   64
#define COUT   2
#define EPSV   1e-5f

typedef __attribute__((ext_vector_type(16))) __bf16 v16bf;
typedef __attribute__((ext_vector_type(8)))  __bf16 v8bf;
typedef __attribute__((ext_vector_type(8)))  float  v8f;

union Frag16 {            // one WMMA 16-bit operand fragment: 8 VGPRs / lane
  v16bf v;
  v8bf  h[2];
};

// ---------------------------------------------------------------------------
// degree = segment_sum(ones, dst) ; then +1, rsqrt, inv
// ---------------------------------------------------------------------------
__global__ void k_deg(const int* __restrict__ dst, float* __restrict__ deg, int E) {
  for (int i = blockIdx.x * blockDim.x + threadIdx.x; i < E; i += gridDim.x * blockDim.x)
    atomicAdd(&deg[dst[i]], 1.0f);
}

__global__ void k_deg_fin(const float* __restrict__ deg, float* __restrict__ dis,
                          float* __restrict__ dinv, int n) {
  for (int i = blockIdx.x * blockDim.x + threadIdx.x; i < n; i += gridDim.x * blockDim.x) {
    float d = deg[i] + 1.0f;
    dis[i]  = rsqrtf(d);
    dinv[i] = 1.0f / d;
  }
}

// ---------------------------------------------------------------------------
// BatchNorm: per-channel sum / sumsq partial reduction
// ---------------------------------------------------------------------------
__global__ void k_bn_stats(const float* __restrict__ h, float* __restrict__ sums,
                           float* __restrict__ sq, int n, int C) {
  int c = threadIdx.x;
  if (c >= C) return;
  float s = 0.0f, s2 = 0.0f;
  for (int r = blockIdx.x; r < n; r += gridDim.x) {
    float v = h[(long)r * C + c];
    s += v; s2 += v * v;
  }
  atomicAdd(&sums[c], s);
  atomicAdd(&sq[c], s2);
}

__global__ void k_bn_fin(const float* __restrict__ sums, const float* __restrict__ sq,
                         const float* __restrict__ g, const float* __restrict__ b,
                         float* __restrict__ scale, float* __restrict__ shift,
                         int C, float invN) {
  int c = threadIdx.x;
  if (c >= C) return;
  float m = sums[c] * invN;
  float v = sq[c] * invN - m * m;
  float a = g[c] * rsqrtf(v + EPSV);
  scale[c] = a;
  shift[c] = b[c] - m * a;
}

__global__ void k_bn_apply(const float* __restrict__ h, const float* __restrict__ scale,
                           const float* __restrict__ shift, float* __restrict__ out,
                           __bf16* __restrict__ outbf, int total, int C) {
  for (int i = blockIdx.x * blockDim.x + threadIdx.x; i < total; i += gridDim.x * blockDim.x) {
    int c = i % C;
    float v = h[i] * scale[c] + shift[c];
    out[i] = v;
    if (outbf) outbf[i] = (__bf16)v;
  }
}

// ---------------------------------------------------------------------------
// Layer-1 GEMM: [N,3] @ [3,128]  (K=3 -> scalar, trivially cheap)
// ---------------------------------------------------------------------------
__global__ void k_gemm3(const float* __restrict__ h0, const float* __restrict__ W,
                        float* __restrict__ hw, int n) {
  for (int i = blockIdx.x * blockDim.x + threadIdx.x; i < n * HID;
       i += gridDim.x * blockDim.x) {
    int r = i / HID, j = i % HID;
    const float* hr = h0 + (long)r * CIN;
    hw[i] = hr[0] * W[j] + hr[1] * W[HID + j] + hr[2] * W[2 * HID + j];
  }
}

// f32 weight [K,NN] row-major -> bf16 transposed K-major Bt[NN,K]
__global__ void k_f2bf_t(const float* __restrict__ W, __bf16* __restrict__ Bt,
                         int K, int NN) {
  for (int i = blockIdx.x * blockDim.x + threadIdx.x; i < K * NN;
       i += gridDim.x * blockDim.x) {
    int n = i / K, k = i % K;
    Bt[i] = (__bf16)W[(long)k * NN + n];
  }
}

// ---------------------------------------------------------------------------
// WMMA GEMM: C[M,NN] = relu?(A[M,128] @ B[128,NN] + bias)
//  - A row-major bf16; Bt = B transposed, K-major bf16 [NN,128]
//  - Block = 256 thr (8 waves). Bt staged in LDS (<=32KB) via b128 loads +
//    ds_store_b128. One wave per 16-row M-tile; A fragments (K=128) loaded
//    once with global_load_b128 and reused across all NN/16 column tiles.
//  - v_wmma_f32_16x16x32_bf16, f32 accumulate, fully unrolled.
// ---------------------------------------------------------------------------
template <int NN>
__global__ void k_wmma_gemm(const __bf16* __restrict__ A, const __bf16* __restrict__ Bt,
                            const float* __restrict__ bias, float* __restrict__ C,
                            int M, int relu) {
  __shared__ __bf16 lds[NN * 128];

  // cooperative stage of Bt into LDS (16B per thread per step)
  for (int i = threadIdx.x * 8; i < NN * 128; i += blockDim.x * 8)
    *(v8bf*)&lds[i] = *(const v8bf*)&Bt[i];
  __syncthreads();

  const int lane = threadIdx.x & 31;
  const int wv   = threadIdx.x >> 5;
  const int tm   = blockIdx.x * (blockDim.x >> 5) + wv;
  if (tm >= (M >> 4)) return;                 // wave-uniform; EXEC stays all-ones

  // ISA 16-bit A 16x32 layout: lanes 0-15 -> K 0-7 / 16-23; lanes 16-31 -> 8-15 / 24-31
  const int mrow = (tm << 4) + (lane & 15);
  const int kbA  = (lane < 16) ? 0 : 8;
  // B 32x16: column = lane & 15; lanes 0-15 hold K 0-7 / 16-23, lanes 16-31 K 8-15 / 24-31
  const int nloc = lane & 15;
  const int kbB  = (lane < 16) ? 0 : 8;

  // preload all 4 A fragments for K=128 (2x global_load_b128 each)
  Frag16 a[4];
  const __bf16* arow = A + (size_t)mrow * 128;
#pragma unroll
  for (int kk = 0; kk < 4; ++kk) {
    a[kk].h[0] = *(const v8bf*)(arow + kk * 32 + kbA);
    a[kk].h[1] = *(const v8bf*)(arow + kk * 32 + 16 + kbA);
  }

  const int rbase = (tm << 4) + ((lane < 16) ? 0 : 8);
#pragma unroll
  for (int tn = 0; tn < NN / 16; ++tn) {
    const __bf16* bcol = &lds[(tn * 16 + nloc) * 128];
    v8f acc = {};
#pragma unroll
    for (int kk = 0; kk < 4; ++kk) {
      Frag16 b;
      b.h[0] = *(const v8bf*)(bcol + kk * 32 + kbB);       // ds_load_b128
      b.h[1] = *(const v8bf*)(bcol + kk * 32 + kbB + 16);  // ds_load_b128
      acc = __builtin_amdgcn_wmma_f32_16x16x32_bf16(false, a[kk].v, false, b.v,
                                                    (short)0, acc, false, false);
    }
    const int ncol = tn * 16 + nloc;
    const float bb = bias ? bias[ncol] : 0.0f;
#pragma unroll
    for (int r = 0; r < 8; ++r) {
      float v = acc[r] + bb;
      if (relu) v = fmaxf(v, 0.0f);
      C[(size_t)(rbase + r) * NN + ncol] = v;
    }
  }
}

// ---------------------------------------------------------------------------
// Edge scatter: agg[dst] += hw[src] * dis[src]*dis[dst]
// 128 threads per edge (one per channel): coalesced 512B gather + f32 atomics.
// ---------------------------------------------------------------------------
__global__ void k_scatter(const int* __restrict__ src, const int* __restrict__ dst,
                          const float* __restrict__ dis, const float* __restrict__ hw,
                          float* __restrict__ agg, int E) {
  const int t   = threadIdx.x & (HID - 1);
  const int epb = blockDim.x >> 7;                    // edges per block-iteration
  int e = blockIdx.x * epb + (threadIdx.x >> 7);
  const int estep = gridDim.x * epb;
  for (; e < E; e += estep) {
    int s = src[e], d = dst[e];
    int e2 = e + estep;                               // stage the next gather row
    if (e2 < E) __builtin_prefetch(&hw[(long)src[e2] * HID + t], 0, 0);
    float coef = dis[s] * dis[d];
    atomicAdd(&agg[(long)d * HID + t], hw[(long)s * HID + t] * coef);
  }
}

// agg = relu(agg + hw * deg_inv + bias)   (in place; pre-BN activation)
__global__ void k_post(float* __restrict__ agg, const float* __restrict__ hw,
                       const float* __restrict__ bias, const float* __restrict__ dinv,
                       int n) {
  for (int i = blockIdx.x * blockDim.x + threadIdx.x; i < n * HID;
       i += gridDim.x * blockDim.x) {
    int r = i / HID, j = i % HID;
    float v = agg[i] + hw[i] * dinv[r] + bias[j];
    agg[i] = fmaxf(v, 0.0f);
  }
}

// ---------------------------------------------------------------------------
// Global mean pool
// ---------------------------------------------------------------------------
__global__ void k_pool(const float* __restrict__ h, const int* __restrict__ batch,
                       float* __restrict__ pooled, float* __restrict__ cnt, int n) {
  const int t   = threadIdx.x & (HID - 1);
  const int npb = blockDim.x >> 7;
  int r = blockIdx.x * npb + (threadIdx.x >> 7);
  for (; r < n; r += gridDim.x * npb) {
    int g = batch[r];
    atomicAdd(&pooled[(long)g * HID + t], h[(long)r * HID + t]);
    if (t == 0) atomicAdd(&cnt[g], 1.0f);
  }
}

__global__ void k_pool_fin(float* __restrict__ pooled, const float* __restrict__ cnt,
                           __bf16* __restrict__ pooledbf, int G) {
  for (int i = blockIdx.x * blockDim.x + threadIdx.x; i < G * HID;
       i += gridDim.x * blockDim.x) {
    int g = i / HID;
    float v = pooled[i] / fmaxf(cnt[g], 1.0f);
    pooled[i]   = v;
    pooledbf[i] = (__bf16)v;
  }
}

// Final tiny GEMM: out[G,2] = z[G,64] @ Wc2[64,2] + bc2
__global__ void k_cls2(const float* __restrict__ z, const float* __restrict__ W,
                       const float* __restrict__ b, float* __restrict__ out, int G) {
  for (int i = blockIdx.x * blockDim.x + threadIdx.x; i < G * COUT;
       i += gridDim.x * blockDim.x) {
    int g = i / COUT, o = i % COUT;
    float acc = b[o];
    for (int j = 0; j < CMID; ++j) acc += z[(long)g * CMID + j] * W[j * COUT + o];
    out[i] = acc;
  }
}

// ---------------------------------------------------------------------------
extern "C" void kernel_launch(void* const* d_in, const int* in_sizes, int n_in,
                              void* d_out, int out_size, void* d_ws, size_t ws_size,
                              hipStream_t stream) {
  const float* x    = (const float*)d_in[0];
  const int*   ei   = (const int*)  d_in[1];
  const int*   bat  = (const int*)  d_in[2];
  const float* W1   = (const float*)d_in[3];
  const float* b1   = (const float*)d_in[4];
  const float* W2   = (const float*)d_in[5];
  const float* b2   = (const float*)d_in[6];
  const float* W3   = (const float*)d_in[7];
  const float* b3   = (const float*)d_in[8];
  const float* bn0g = (const float*)d_in[9],  *bn0b = (const float*)d_in[10];
  const float* bn1g = (const float*)d_in[11], *bn1b = (const float*)d_in[12];
  const float* bn2g = (const float*)d_in[13], *bn2b = (const float*)d_in[14];
  const float* bn3g = (const float*)d_in[15], *bn3b = (const float*)d_in[16];
  const float* Wc1  = (const float*)d_in[17];
  const float* bc1  = (const float*)d_in[18];
  const float* Wc2  = (const float*)d_in[19];
  const float* bc2  = (const float*)d_in[20];
  float* out = (float*)d_out;

  const int N = NNODES, E = NEDGES, G = NGRAPH;
  const int* src = ei;
  const int* dst = ei + E;

  // workspace carve-up (256B aligned)
  char* base = (char*)d_ws;
  size_t off = 0;
  auto carve = [&](size_t bytes) -> void* {
    off = (off + 255) & ~(size_t)255;
    void* p = base + off;
    off += bytes;
    return p;
  };
  float*  deg   = (float*)carve((size_t)N * 4);
  float*  dis   = (float*)carve((size_t)N * 4);
  float*  dinv  = (float*)carve((size_t)N * 4);
  float*  h0    = (float*)carve((size_t)N * CIN * 4);
  float*  h     = (float*)carve((size_t)N * HID * 4);
  __bf16* hbf   = (__bf16*)carve((size_t)N * HID * 2);
  float*  hw    = (float*)carve((size_t)N * HID * 4);
  float*  agg   = (float*)carve((size_t)N * HID * 4);
  float*  sums  = (float*)carve(HID * 4);
  float*  sq    = (float*)carve(HID * 4);
  float*  scale = (float*)carve(HID * 4);
  float*  shift = (float*)carve(HID * 4);
  __bf16* w2bf  = (__bf16*)carve(HID * HID * 2);   // transposed, K-major
  __bf16* w3bf  = (__bf16*)carve(HID * HID * 2);   // transposed, K-major
  __bf16* wc1bf = (__bf16*)carve(HID * CMID * 2);  // transposed, K-major
  float*  pooled= (float*)carve((size_t)G * HID * 4);
  float*  cnt   = (float*)carve((size_t)G * 4);
  __bf16* plbf  = (__bf16*)carve((size_t)G * HID * 2);
  float*  z     = (float*)carve((size_t)G * CMID * 4);

  const int T = 256;
  auto nb = [&](long n) { return (int)((n + T - 1) / T); };
  const int NH = N * HID;

  // degree
  hipMemsetAsync(deg, 0, (size_t)N * 4, stream);
  k_deg<<<2048, T, 0, stream>>>(dst, deg, E);
  k_deg_fin<<<nb(N), T, 0, stream>>>(deg, dis, dinv, N);

  // weight conversions to bf16 (transposed K-major)
  k_f2bf_t<<<nb(HID * HID), T, 0, stream>>>(W2, w2bf, HID, HID);
  k_f2bf_t<<<nb(HID * HID), T, 0, stream>>>(W3, w3bf, HID, HID);
  k_f2bf_t<<<nb(HID * CMID), T, 0, stream>>>(Wc1, wc1bf, HID, CMID);

  // BN0 on x -> h0
  hipMemsetAsync(sums, 0, HID * 4, stream);
  hipMemsetAsync(sq,   0, HID * 4, stream);
  k_bn_stats<<<512, HID, 0, stream>>>(x, sums, sq, N, CIN);
  k_bn_fin<<<1, HID, 0, stream>>>(sums, sq, bn0g, bn0b, scale, shift, CIN, 1.0f / N);
  k_bn_apply<<<nb((long)N * CIN), T, 0, stream>>>(x, scale, shift, h0, nullptr, N * CIN, CIN);

  // ---- layer 1 (K=3 scalar GEMM) ----
  k_gemm3<<<nb(NH), T, 0, stream>>>(h0, W1, hw, N);
  hipMemsetAsync(agg, 0, (size_t)NH * 4, stream);
  k_scatter<<<8192, T, 0, stream>>>(src, dst, dis, hw, agg, E);
  k_post<<<nb(NH), T, 0, stream>>>(agg, hw, b1, dinv, N);
  hipMemsetAsync(sums, 0, HID * 4, stream);
  hipMemsetAsync(sq,   0, HID * 4, stream);
  k_bn_stats<<<512, HID, 0, stream>>>(agg, sums, sq, N, HID);
  k_bn_fin<<<1, HID, 0, stream>>>(sums, sq, bn1g, bn1b, scale, shift, HID, 1.0f / N);
  k_bn_apply<<<nb(NH), T, 0, stream>>>(agg, scale, shift, h, hbf, NH, HID);

  const int mblocks = ((N >> 4) + 7) / 8;   // one wave per 16-row tile, 8 waves/block

  // ---- layer 2 (WMMA bf16 GEMM) ----
  k_wmma_gemm<HID><<<mblocks, T, 0, stream>>>(hbf, w2bf, nullptr, hw, N, 0);
  hipMemsetAsync(agg, 0, (size_t)NH * 4, stream);
  k_scatter<<<8192, T, 0, stream>>>(src, dst, dis, hw, agg, E);
  k_post<<<nb(NH), T, 0, stream>>>(agg, hw, b2, dinv, N);
  hipMemsetAsync(sums, 0, HID * 4, stream);
  hipMemsetAsync(sq,   0, HID * 4, stream);
  k_bn_stats<<<512, HID, 0, stream>>>(agg, sums, sq, N, HID);
  k_bn_fin<<<1, HID, 0, stream>>>(sums, sq, bn2g, bn2b, scale, shift, HID, 1.0f / N);
  k_bn_apply<<<nb(NH), T, 0, stream>>>(agg, scale, shift, h, hbf, NH, HID);

  // ---- layer 3 (WMMA bf16 GEMM) ----
  k_wmma_gemm<HID><<<mblocks, T, 0, stream>>>(hbf, w3bf, nullptr, hw, N, 0);
  hipMemsetAsync(agg, 0, (size_t)NH * 4, stream);
  k_scatter<<<8192, T, 0, stream>>>(src, dst, dis, hw, agg, E);
  k_post<<<nb(NH), T, 0, stream>>>(agg, hw, b3, dinv, N);
  hipMemsetAsync(sums, 0, HID * 4, stream);
  hipMemsetAsync(sq,   0, HID * 4, stream);
  k_bn_stats<<<512, HID, 0, stream>>>(agg, sums, sq, N, HID);
  k_bn_fin<<<1, HID, 0, stream>>>(sums, sq, bn3g, bn3b, scale, shift, HID, 1.0f / N);
  k_bn_apply<<<nb(NH), T, 0, stream>>>(agg, scale, shift, h, hbf, NH, HID);

  // ---- pool ----
  hipMemsetAsync(pooled, 0, (size_t)G * HID * 4, stream);
  hipMemsetAsync(cnt,    0, (size_t)G * 4, stream);
  k_pool<<<4096, T, 0, stream>>>(h, bat, pooled, cnt, N);
  k_pool_fin<<<nb((long)G * HID), T, 0, stream>>>(pooled, cnt, plbf, G);

  // ---- classifier: [512,128]@[128,64] via WMMA, then [512,64]@[64,2] scalar ----
  k_wmma_gemm<CMID><<<((G >> 4) + 7) / 8, T, 0, stream>>>(plbf, wc1bf, bc1, z, G, 1);
  k_cls2<<<nb((long)G * COUT), T, 0, stream>>>(z, Wc2, bc2, out, G);
}
